// InsideEncoder_8727373545739
// MI455X (gfx1250) — matrix operand order repaired and checked by reading, not compiled, not run on hardware
//
#include <hip/hip_runtime.h>

// ---------------------------------------------------------------------------
// PointNet-style segment encoder for MI455X (gfx1250, wave32, WMMA).
//
// Pipeline (all on `stream`):
//   0. zero_kernel      : zero rel_enc[B,32] and pooled[B,128] in d_ws
//   1. convert_weights  : fp32 weights -> transposed f16 [Nout][K] in d_ws
//   2. encoder_kernel   : enc = relu(points @ w_ne + b_ne); segment-max ->
//                         rel_enc via uint atomicMax (relu => >= 0)
//   3. point_mlp_kernel : per-point 3-layer MLP fused with
//                         v_wmma_f32_16x16x32_f16; 64 points (4 M-tiles) per
//                         wave so each B fragment is reused by 4 WMMAs;
//                         segment-max -> pooled with run-length-flushed atomics
//   4. seg_mlp_kernel   : [B,128] -> 128 -> 64 -> 32 via WMMA, write d_out
// ---------------------------------------------------------------------------

#define N_PTS 1048576
#define BSEG  2048
#define FEAT  64
#define ENC   32
#define KD    128   // padded K stride (halves) for activation rows in LDS

typedef __fp16   f16x2 __attribute__((ext_vector_type(2)));
typedef __fp16   f16x4 __attribute__((ext_vector_type(4)));
typedef _Float16 h8    __attribute__((ext_vector_type(8)));
typedef _Float16 v16h  __attribute__((ext_vector_type(16)));
typedef float    v8f   __attribute__((ext_vector_type(8)));

// Single-instruction ReLU. The IR-level max (fmaxf / fmed3 builtin) gets
// canonicalized into two v_max_num ops; raw asm bypasses that.
static __device__ __forceinline__ float relu1(float x) {
    float r;
    asm("v_max_num_f32 %0, %1, 0" : "=v"(r) : "v"(x));
    return r;
}

// float4 -> 4x f16 via packed converts (v_cvt_pk_rtz_f16_f32).
static __device__ __forceinline__ f16x4 cvt4(float4 f) {
    f16x2 lo = __builtin_amdgcn_cvt_pkrtz(f.x, f.y);
    f16x2 hi = __builtin_amdgcn_cvt_pkrtz(f.z, f.w);
    return __builtin_shufflevector(lo, hi, 0, 1, 2, 3);
}

// ---- WMMA fragment loaders (wave32, contiguous k order, same for A and B) ----------

// A 16x32 f16: lane L -> row M=L%16; halves K = 16*(L/16)+{0..15}.
static __device__ __forceinline__ v16h load_a_frag(const _Float16* act, int lane, int kstep) {
    int m  = lane & 15;
    int kb = ((lane >> 4) << 4) + (kstep << 5);
    h8 lo = *(const h8*)(act + m * KD + kb);
    h8 hi = *(const h8*)(act + m * KD + kb + 8);
    return __builtin_shufflevector(lo, hi, 0,1,2,3,4,5,6,7,8,9,10,11,12,13,14,15);
}

// B 32x16 f16: lane L -> col N=L%16; halves K = 16*(L/16)+{0..15}.
// wT is [Nout][Kdim] row-major f16 (transposed weights), so this is 2x16B.
static __device__ __forceinline__ v16h load_b_frag(const _Float16* wT, int kdim,
                                                   int lane, int ntile, int kstep) {
    int n  = (ntile << 4) + (lane & 15);
    int kb = ((lane >> 4) << 4) + (kstep << 5);
    const h8* q = (const h8*)(wT + n * kdim + kb);
    h8 lo = q[0];
    h8 hi = q[1];
    return __builtin_shufflevector(lo, hi, 0,1,2,3,4,5,6,7,8,9,10,11,12,13,14,15);
}

// One dense layer on MTILES*16 rows, in place in LDS:
//   act[0:MTILES*16, 0:KDIM] -> relu(act @ W + b) -> act[0:MTILES*16, 0:NOUT]
// All A fragments are loaded into registers before the write-back, so a
// single buffer is safe. Each B fragment is reused by MTILES WMMAs.
template <int KDIM, int NOUT, int MTILES>
static __device__ __forceinline__ void mlp_layer(_Float16* act,
                                                 const _Float16* __restrict__ wT,
                                                 const float* __restrict__ bias,
                                                 int lane) {
    constexpr int KS = KDIM / 32;
    constexpr int NT = NOUT / 16;
    v16h afrag[MTILES * KS];
#pragma unroll
    for (int mt = 0; mt < MTILES; ++mt)
#pragma unroll
        for (int k = 0; k < KS; ++k)
            afrag[mt * KS + k] = load_a_frag(act + mt * 16 * KD, lane, k);

    const int ncol = lane & 15;
    const int mrow = (lane >> 4) << 3;
#pragma unroll
    for (int nt = 0; nt < NT; ++nt) {
        float bn = bias[(nt << 4) + ncol];
        v8f acc[MTILES];
#pragma unroll
        for (int mt = 0; mt < MTILES; ++mt)
#pragma unroll
            for (int r = 0; r < 8; ++r) acc[mt][r] = bn;
#pragma unroll
        for (int k = 0; k < KS; ++k) {
            v16h bf = load_b_frag(wT, KDIM, lane, nt, k);
#pragma unroll
            for (int mt = 0; mt < MTILES; ++mt)
                acc[mt] = __builtin_amdgcn_wmma_f32_16x16x32_f16(
                    false, afrag[mt * KS + k], false, bf, (short)0, acc[mt], false, false);
        }
        int col = (nt << 4) + ncol;
#pragma unroll
        for (int mt = 0; mt < MTILES; ++mt)
#pragma unroll
            for (int r = 0; r < 8; ++r) {
                act[(mt * 16 + mrow + r) * KD + col] = (_Float16)relu1(acc[mt][r]);
            }
    }
}

// ---- 0) zero scratch ------------------------------------------------------
__global__ void zero_kernel(float* __restrict__ p, int n) {
    int i = blockIdx.x * blockDim.x + threadIdx.x;
    if (i < n) p[i] = 0.f;
}

// ---- 1) fp32 weights -> f16 transposed [Nout][K] --------------------------
__global__ void convert_weights_kernel(const float* __restrict__ w1,  const float* __restrict__ w2,
                                       const float* __restrict__ w3,  const float* __restrict__ wg1,
                                       const float* __restrict__ wg2, const float* __restrict__ wg3,
                                       _Float16* __restrict__ w1t,  _Float16* __restrict__ w2t,
                                       _Float16* __restrict__ w3t,  _Float16* __restrict__ wg1t,
                                       _Float16* __restrict__ wg2t, _Float16* __restrict__ wg3t) {
    int i = blockIdx.x * blockDim.x + threadIdx.x;
    if (i < 128 * 96)  { int n = i / 96,  k = i % 96;  w1t[i]  = (_Float16)w1[k * 128 + n]; return; }
    i -= 128 * 96;
    if (i < 128 * 128) { int n = i / 128, k = i % 128; w2t[i]  = (_Float16)w2[k * 128 + n]; return; }
    i -= 128 * 128;
    if (i < 128 * 128) { int n = i / 128, k = i % 128; w3t[i]  = (_Float16)w3[k * 128 + n]; return; }
    i -= 128 * 128;
    if (i < 128 * 128) { int n = i / 128, k = i % 128; wg1t[i] = (_Float16)wg1[k * 128 + n]; return; }
    i -= 128 * 128;
    if (i < 64 * 128)  { int n = i / 128, k = i % 128; wg2t[i] = (_Float16)wg2[k * 64 + n]; return; }
    i -= 64 * 128;
    if (i < 32 * 64)   { int n = i / 64,  k = i % 64;  wg3t[i] = (_Float16)wg3[k * 32 + n]; return; }
}

// ---- 2) neighborhood encoder + segment-max (K=3 -> VALU) ------------------
// Wave handles 32 points x 32 enc-cols; lane owns one column. Sorted batch
// lets us keep a running max and flush one atomic per segment run.
__global__ void encoder_kernel(const float* __restrict__ points,
                               const int*   __restrict__ batch,
                               const float* __restrict__ w_ne,
                               const float* __restrict__ b_ne,
                               float* __restrict__ rel_enc) {
    __shared__ float pts[8][32][3];
    __shared__ int   pb[8][32];
    int wave = threadIdx.x >> 5;
    int lane = threadIdx.x & 31;
    int base = (blockIdx.x * 8 + wave) * 32;

    pts[wave][lane][0] = points[(base + lane) * 3 + 0];
    pts[wave][lane][1] = points[(base + lane) * 3 + 1];
    pts[wave][lane][2] = points[(base + lane) * 3 + 2];
    pb[wave][lane]     = batch[base + lane];
    __syncthreads();

    float wx = w_ne[0 * ENC + lane];
    float wy = w_ne[1 * ENC + lane];
    float wz = w_ne[2 * ENC + lane];
    float bj = b_ne[lane];

    unsigned run = 0u;  // bits of 0.0f
#pragma unroll
    for (int p = 0; p < 32; ++p) {
        float e = relu1(pts[wave][p][0] * wx + pts[wave][p][1] * wy + pts[wave][p][2] * wz + bj);
        unsigned ue = __float_as_uint(e);
        run = run > ue ? run : ue;
        if (p == 31 || pb[wave][p + 1] != pb[wave][p]) {
            atomicMax((unsigned*)rel_enc + pb[wave][p] * ENC + lane, run);
            run = 0u;
        }
    }
}

// ---- 3) fused per-point 3-layer MLP + segment-max pool --------------------
// 2 waves/block; each wave owns 64 points (4 M-tiles) in one in-place LDS
// activation buffer (64 x 128 f16 = 16 KB/wave).
__global__ void __launch_bounds__(64)
point_mlp_kernel(const float* __restrict__ rel_enc,
                 const float* __restrict__ features,
                 const int*   __restrict__ batch,
                 const _Float16* __restrict__ w1t, const float* __restrict__ b1,
                 const _Float16* __restrict__ w2t, const float* __restrict__ b2,
                 const _Float16* __restrict__ w3t, const float* __restrict__ b3,
                 float* __restrict__ pooled) {
    __shared__ _Float16 buf[2][64 * KD];
    __shared__ int      bb[2][64];

    int wave = threadIdx.x >> 5;
    int lane = threadIdx.x & 31;
    int base = (blockIdx.x * 2 + wave) * 64;
    _Float16* act = buf[wave];
    int* bbw = bb[wave];

    for (int i = lane; i < 64; i += 32) bbw[i] = batch[base + i];
    __syncthreads();

    // Stage fc_input = [rel_enc[batch], features] as f16 rows [64 x 96],
    // vectorized: float4 global loads -> packed cvt -> 8B LDS stores.
    for (int idx = lane; idx < 64 * 8; idx += 32) {         // rel_enc: 8 float4/row
        int m = idx >> 3, c4 = idx & 7;
        float4 f = ((const float4*)(rel_enc + bbw[m] * ENC))[c4];
        *(f16x4*)(act + m * KD + c4 * 4) = cvt4(f);
    }
    for (int idx = lane; idx < 64 * 16; idx += 32) {        // features: 16 float4/row
        int m = idx >> 4, c4 = idx & 15;
        float4 f = ((const float4*)(features + (base + m) * FEAT))[c4];
        *(f16x4*)(act + m * KD + ENC + c4 * 4) = cvt4(f);
    }

    mlp_layer<96, 128, 4>(act, w1t, b1, lane);    // layer 1
    mlp_layer<128, 128, 4>(act, w2t, b2, lane);   // layer 2

    // Layer 3 fused with segment-max pooling (run-length flushed atomics).
    v16h afrag[16];
#pragma unroll
    for (int mt = 0; mt < 4; ++mt)
#pragma unroll
        for (int k = 0; k < 4; ++k)
            afrag[mt * 4 + k] = load_a_frag(act + mt * 16 * KD, lane, k);

    const int ncol = lane & 15;
    const int mrow = (lane >> 4) << 3;
#pragma unroll
    for (int nt = 0; nt < 8; ++nt) {
        float bn = b3[(nt << 4) + ncol];
        v8f acc[4];
#pragma unroll
        for (int mt = 0; mt < 4; ++mt)
#pragma unroll
            for (int r = 0; r < 8; ++r) acc[mt][r] = bn;
#pragma unroll
        for (int k = 0; k < 4; ++k) {
            v16h bf = load_b_frag(w3t, 128, lane, nt, k);
#pragma unroll
            for (int mt = 0; mt < 4; ++mt)
                acc[mt] = __builtin_amdgcn_wmma_f32_16x16x32_f16(
                    false, afrag[mt * 4 + k], false, bf, (short)0, acc[mt], false, false);
        }
        int col = (nt << 4) + ncol;
#pragma unroll
        for (int mt = 0; mt < 4; ++mt) {
            unsigned run = 0u;
#pragma unroll
            for (int r = 0; r < 8; ++r) {
                unsigned uv = __float_as_uint(relu1(acc[mt][r]));
                run = run > uv ? run : uv;
                int m = mt * 16 + mrow + r;
                if (r == 7 || bbw[m + 1] != bbw[m]) {   // short-circuit guards OOB
                    atomicMax((unsigned*)pooled + bbw[m] * 128 + col, run);
                    run = 0u;
                }
            }
        }
    }
}

// ---- 4) tail MLP on pooled [B,128] -> [B,32] ------------------------------
// Same 4-M-tile scheme: 2 waves/block, 64 segments/wave.
__global__ void __launch_bounds__(64)
seg_mlp_kernel(const float* __restrict__ pooled,
               const _Float16* __restrict__ wg1t, const float* __restrict__ bg1,
               const _Float16* __restrict__ wg2t, const float* __restrict__ bg2,
               const _Float16* __restrict__ wg3t, const float* __restrict__ bg3,
               float* __restrict__ out) {
    __shared__ _Float16 buf[2][64 * KD];
    int wave = threadIdx.x >> 5;
    int lane = threadIdx.x & 31;
    int base = (blockIdx.x * 2 + wave) * 64;
    _Float16* act = buf[wave];

    for (int idx = lane; idx < 64 * 32; idx += 32) {        // pooled: 32 float4/row
        int m = idx >> 5, c4 = idx & 31;
        float4 f = ((const float4*)(pooled + (base + m) * 128))[c4];
        *(f16x4*)(act + m * KD + c4 * 4) = cvt4(f);
    }

    mlp_layer<128, 128, 4>(act, wg1t, bg1, lane);  // g1
    mlp_layer<128, 64, 4>(act, wg2t, bg2, lane);   // g2 (writes cols 0..63)

    // g3: K=64, Nout=32, f32 output straight to d_out.
    v16h afrag[8];
#pragma unroll
    for (int mt = 0; mt < 4; ++mt)
#pragma unroll
        for (int k = 0; k < 2; ++k)
            afrag[mt * 2 + k] = load_a_frag(act + mt * 16 * KD, lane, k);

    const int ncol = lane & 15;
    const int mrow = (lane >> 4) << 3;
#pragma unroll
    for (int nt = 0; nt < 2; ++nt) {
        float bn = bg3[(nt << 4) + ncol];
        v8f acc[4];
#pragma unroll
        for (int mt = 0; mt < 4; ++mt)
#pragma unroll
            for (int r = 0; r < 8; ++r) acc[mt][r] = bn;
#pragma unroll
        for (int k = 0; k < 2; ++k) {
            v16h bf = load_b_frag(wg3t, 64, lane, nt, k);
#pragma unroll
            for (int mt = 0; mt < 4; ++mt)
                acc[mt] = __builtin_amdgcn_wmma_f32_16x16x32_f16(
                    false, afrag[mt * 2 + k], false, bf, (short)0, acc[mt], false, false);
        }
        int col = (nt << 4) + ncol;
#pragma unroll
        for (int mt = 0; mt < 4; ++mt)
#pragma unroll
            for (int r = 0; r < 8; ++r) {
                out[(base + mt * 16 + mrow + r) * 32 + col] = relu1(acc[mt][r]);
            }
    }
}

// ---------------------------------------------------------------------------
extern "C" void kernel_launch(void* const* d_in, const int* in_sizes, int n_in,
                              void* d_out, int out_size, void* d_ws, size_t ws_size,
                              hipStream_t stream) {
    const float* points   = (const float*)d_in[0];
    const float* features = (const float*)d_in[1];
    const int*   batch    = (const int*)d_in[2];   // sorted segment ids
    const float* w_ne = (const float*)d_in[3];
    const float* b_ne = (const float*)d_in[4];
    const float* w1   = (const float*)d_in[5];
    const float* b1   = (const float*)d_in[6];
    const float* w2   = (const float*)d_in[7];
    const float* b2   = (const float*)d_in[8];
    const float* w3   = (const float*)d_in[9];
    const float* b3   = (const float*)d_in[10];
    const float* wg1  = (const float*)d_in[11];
    const float* bg1  = (const float*)d_in[12];
    const float* wg2  = (const float*)d_in[13];
    const float* bg2  = (const float*)d_in[14];
    const float* wg3  = (const float*)d_in[15];
    const float* bg3  = (const float*)d_in[16];

    // Workspace layout (~1.42 MB total):
    char* ws = (char*)d_ws;
    float* rel_enc = (float*)ws;                       // 2048*32  f32 (256 KB)
    float* pooled  = (float*)(ws + 262144);            // 2048*128 f32 (1 MB)
    _Float16* w1t  = (_Float16*)(ws + 1310720);        // 128*96  f16
    _Float16* w2t  = w1t  + 128 * 96;                  // 128*128 f16
    _Float16* w3t  = w2t  + 128 * 128;
    _Float16* wg1t = w3t  + 128 * 128;
    _Float16* wg2t = wg1t + 128 * 128;
    _Float16* wg3t = wg2t + 64 * 128;

    // 0) zero rel_enc + pooled (contiguous: 2048*(32+128) floats)
    {
        int n = BSEG * (ENC + 128);
        zero_kernel<<<(n + 255) / 256, 256, 0, stream>>>(rel_enc, n);
    }
    // 1) weight conversion (71680 elements total)
    {
        int n = 128 * 96 + 3 * 128 * 128 + 64 * 128 + 32 * 64;
        convert_weights_kernel<<<(n + 255) / 256, 256, 0, stream>>>(
            w1, w2, w3, wg1, wg2, wg3, w1t, w2t, w3t, wg1t, wg2t, wg3t);
    }
    // 2) encoder + first segment-max: 256 points/block (8 waves x 32 points)
    encoder_kernel<<<N_PTS / 256, 256, 0, stream>>>(points, batch, w_ne, b_ne, rel_enc);
    // 3) fused per-point MLP: 128 points/block (2 waves x 64 points)
    point_mlp_kernel<<<N_PTS / 128, 64, 0, stream>>>(
        rel_enc, features, batch, w1t, b1, w2t, b2, w3t, b3, pooled);
    // 4) tail MLP: 128 segments/block
    seg_mlp_kernel<<<BSEG / 128, 64, 0, stream>>>(
        pooled, wg1t, bg1, wg2t, bg2, wg3t, bg3, (float*)d_out);
}
